// RelationalGraphConvLayer_5995774345549
// MI455X (gfx1250) — matrix operand order repaired and checked by reading, not compiled, not run on hardware
//
#include <hip/hip_runtime.h>
#include <hip/hip_bf16.h>

typedef __attribute__((ext_vector_type(16))) __bf16       v16bf;
typedef __attribute__((ext_vector_type(8)))  float        v8f;
typedef __attribute__((ext_vector_type(4)))  float        f32x4;
typedef __attribute__((ext_vector_type(4)))  unsigned int u32x4;

union Frag { u32x4 q[2]; v16bf h; };

// fp32 -> bf16 bits, round-to-nearest-even
__device__ __forceinline__ unsigned int bfbits(float f) {
  unsigned int u = __builtin_bit_cast(unsigned int, f);
  return (u + 0x7FFFu + ((u >> 16) & 1u)) >> 16;
}
__device__ __forceinline__ unsigned int pk2(float lo, float hi) {
  return bfbits(lo) | (bfbits(hi) << 16);
}

__device__ __forceinline__ void sched_fence() {
#if __has_builtin(__builtin_amdgcn_sched_barrier)
  __builtin_amdgcn_sched_barrier(0);   // keep fragment loads above the WMMAs
#endif
}

// Workgroup barrier WITHOUT the global fence __syncthreads() implies.
// Producer side drains LDS stores (dscnt) and async global->LDS (asynccnt);
// plain global prefetch loads may stay in flight across the barrier.
__device__ __forceinline__ void block_sync() {
#if __has_builtin(__builtin_amdgcn_s_barrier_signal) && __has_builtin(__builtin_amdgcn_s_barrier_wait)
  asm volatile("s_wait_dscnt 0\n\ts_wait_asynccnt 0" ::: "memory");
  __builtin_amdgcn_s_barrier_signal(-1);
  __builtin_amdgcn_s_barrier_wait((short)-1);
  asm volatile("" ::: "memory");
#else
  __syncthreads();
#endif
}

// Async copy 16B global -> LDS (per-lane scatter), tracked by ASYNCcnt.
// lds_off is the integer value of an addrspace(3) pointer (= LDS byte offset).
__device__ __forceinline__ void async_g2l_b128(unsigned lds_off, const void* gaddr) {
  asm volatile("global_load_async_to_lds_b128 %0, %1, off"
               :: "v"(lds_off), "v"(gaddr) : "memory");
}

// ---------------------------------------------------------------------------
// K0: Wc[r*256+i][o] = sum_b w_rel[r,b] * w_bases[b,i,o]   (bf16, [2048][256])
// ---------------------------------------------------------------------------
__global__ __launch_bounds__(256)
void k_combine_w(const float* __restrict__ w_bases, const float* __restrict__ w_rel,
                 unsigned short* __restrict__ Wc) {
  const int kk = blockIdx.x;      // 0..2047
  const int o  = threadIdx.x;     // 0..255
  const int r = kk >> 8, i = kk & 255;
  float acc = 0.f;
#pragma unroll
  for (int b = 0; b < 4; ++b)
    acc += w_rel[r * 4 + b] * w_bases[((size_t)b * 256 + i) * 256 + o];
  Wc[(size_t)kk * 256 + o] = (unsigned short)bfbits(acc);
}

// ---------------------------------------------------------------------------
// K1: Yt[o][m] = sum_kk Xmat[m][kk] * Wc[kk][o], Xmat[m][r*256+i] = x[m,i,r]
//     Yt stored TRANSPOSED [256][8192] bf16 (packed dword pairs along m).
// ---------------------------------------------------------------------------
__global__ __launch_bounds__(256)
void k_make_yt(const float* __restrict__ x, const unsigned short* __restrict__ Wc,
               unsigned int* __restrict__ Yt32) {
  __shared__ __align__(16) float lx[4 * 2048];   // 32 KB: 4 rows in kk-order
  const int t  = threadIdx.x;
  const int m0 = blockIdx.x * 4;
  const float* xbase = x + (size_t)m0 * 2048;
  // reorder memory index p = i*8+r  ->  kk = (p&7)*256 + (p>>3)
  for (int it = 0; it < 8; ++it) {
    int linear = it * 1024 + t * 4;
    int row = linear >> 11;
    int p   = linear & 2047;
    f32x4 v = *(const f32x4*)(xbase + (size_t)row * 2048 + p);
#pragma unroll
    for (int j = 0; j < 4; ++j) {
      int pj = p + j;
      lx[row * 2048 + ((pj & 7) << 8) + (pj >> 3)] = v[j];
    }
  }
  __syncthreads();
  float a0 = 0.f, a1 = 0.f, a2 = 0.f, a3 = 0.f;
  for (int kk = 0; kk < 2048; ++kk) {
    float w = __builtin_bit_cast(float, (unsigned int)Wc[(size_t)kk * 256 + t] << 16);
    a0 += lx[0 * 2048 + kk] * w;
    a1 += lx[1 * 2048 + kk] * w;
    a2 += lx[2 * 2048 + kk] * w;
    a3 += lx[3 * 2048 + kk] * w;
  }
  size_t base = ((size_t)t * 8192 + m0) >> 1;   // dword units
  Yt32[base]     = pk2(a0, a1);
  Yt32[base + 1] = pk2(a2, a3);
}

// ---------------------------------------------------------------------------
// Main: Out[8192,256] = A[8192,8192](f32->bf16, converted in-regs) @ Y(bf16)
// Block: 256 thr = 8 waves; tile 32(M) x 256(N) -> 256 workgroups, A read once.
// Wave: 16(M) x 64(N) = 4 accumulators. K-step 64 (two WMMA k-halves).
// A: global f32 -> regs -> bf16 pack -> ds_store_b128 (fragment-major LDS).
// B: global bf16 -> LDS directly via global_load_async_to_lds_b128 (ASYNCcnt).
// LDS double-buffered; ONE fence-free barrier per stage (dscnt+asynccnt drain).
// A sched_barrier splits fragment ds_loads from the WMMA block so waitcnts
// stagger (no per-WMMA s_wait_dscnt 0).
// ---------------------------------------------------------------------------
__global__ __launch_bounds__(256)
void k_main_gemm(const float* __restrict__ A, const unsigned int* __restrict__ Yt32,
                 float* __restrict__ Out) {
  __shared__ __align__(16) unsigned char ldsA[2 * 4096];    // buf x (2 khalf x 2 strip x 1KB)
  __shared__ __align__(16) unsigned char ldsB[2 * 32768];   // buf x (2 khalf x 16 ntile x 1KB)

  const int t     = threadIdx.x;
  const int lane  = t & 31;
  const int wave  = t >> 5;
  const int strip = wave & 1;   // M sub-strip (16 rows)
  const int nq    = wave >> 1;  // N quarter (64 cols = 4 tiles)
  const size_t m0 = (size_t)blockIdx.x * 32;

  // cooperative A-load: thread -> (row = t>>3, chunk ac of 8 floats, k = 8*ac)
  const int arow = t >> 3;
  const int ac   = t & 7;
  const int ac4  = ac & 3;
  const float* aptr = A + (m0 + (size_t)arow) * 8192 + ac * 8;
  const int aoff = ((ac >> 2) << 11)                          // khalf * 2048
                 + ((arow >> 4) << 10)                        // strip * 1024
                 + (((arow & 15) + ((ac4 & 1) << 4)) << 5)    // lane * 32
                 + ((ac4 >> 1) << 4);                         // j-quad * 16

  // cooperative B: thread t = column n of Yt; 64 bf16 (= 128 B) per stage,
  // 8 async b128 copies straight into fragment-major LDS positions.
  const char* bptr = (const char*)(Yt32 + (size_t)t * 4096);
  const unsigned bbase = (unsigned)(size_t)(ldsB
                       + ((t >> 4) << 10)            // ntile * 1024
                       + ((t & 15) << 5));           // lane-low * 32

  v8f acc[4];
#pragma unroll
  for (int i = 0; i < 4; ++i)
#pragma unroll
    for (int j = 0; j < 8; ++j) acc[i][j] = 0.f;

  f32x4 ra0, ra1;

  const int KIT = 8192 / 64;   // 128 stages

  auto loadA = [&](int it) {
    const float* ap = aptr + (size_t)it * 64;
    ra0 = *(const f32x4*)(ap);
    ra1 = *(const f32x4*)(ap + 4);
  };

  auto storeA = [&](int buf) {
    u32x4 apk;
    apk[0] = pk2(ra0[0], ra0[1]); apk[1] = pk2(ra0[2], ra0[3]);
    apk[2] = pk2(ra1[0], ra1[1]); apk[3] = pk2(ra1[2], ra1[3]);
    *(u32x4*)(ldsA + buf * 4096 + aoff) = apk;        // ds_store_b128
  };

  auto asyncB = [&](int it, int buf) {
    const char* g = bptr + (size_t)it * 128;
    const unsigned lb = bbase + buf * 32768;
#pragma unroll
    for (int c2 = 0; c2 < 8; ++c2) {
      // fragment offset: khalf*16384 + lane-high-bit*512 + j-quad*16
      const unsigned C = ((c2 & 4) << 12) + ((c2 & 1) << 9) + (((c2 >> 1) & 1) << 4);
      async_g2l_b128(lb + C, g + c2 * 16);
    }
  };

  auto compute = [&](int buf) {
    const unsigned char* ab = ldsA + buf * 4096 + strip * 1024 + lane * 32;
    const unsigned char* bb = ldsB + buf * 32768 + nq * 4096 + lane * 32;
    Frag af[2];
    Frag bf[8];
#pragma unroll
    for (int kh = 0; kh < 2; ++kh) {
      const u32x4* pa = (const u32x4*)(ab + kh * 2048);
      af[kh].q[0] = pa[0]; af[kh].q[1] = pa[1];
#pragma unroll
      for (int nt = 0; nt < 4; ++nt) {
        const u32x4* pb = (const u32x4*)(bb + kh * 16384 + nt * 1024);
        bf[kh * 4 + nt].q[0] = pb[0];
        bf[kh * 4 + nt].q[1] = pb[1];
      }
    }
    sched_fence();   // all 20 ds_loads issued before first WMMA -> staggered waits
#pragma unroll
    for (int kh = 0; kh < 2; ++kh)
#pragma unroll
      for (int nt = 0; nt < 4; ++nt)
        acc[nt] = __builtin_amdgcn_wmma_f32_16x16x32_bf16(
            false, af[kh].h, false, bf[kh * 4 + nt].h, (short)0, acc[nt], false, false);
  };

  // prologue: stage 0 -> LDS buf0 (async B + packed A), stage 1 A in registers
  asyncB(0, 0);
  loadA(0);
  storeA(0);
  loadA(1);

  for (int it = 0; it < KIT; ++it) {
    block_sync();                                   // buf[it&1] now complete/visible
    if (it + 1 < KIT) {
      asyncB(it + 1, (it + 1) & 1);                 // global->LDS, no registers
      storeA((it + 1) & 1);                         // A regs loaded 1 iter ago
    }
    if (it + 2 < KIT) loadA(it + 2);                // overlaps compute below
    if (it + 8 < KIT) __builtin_prefetch(aptr + (size_t)(it + 8) * 64, 0, 3);
    compute(it & 1);
  }

  // C/D layout: lanes 0-15: M=v, N=lane; lanes 16-31: M=8+v, N=lane-16
  const int orow = (lane >> 4) * 8;
  const int ocol = lane & 15;
#pragma unroll
  for (int nt = 0; nt < 4; ++nt) {
    float* op = Out + (m0 + strip * 16 + orow) * 256 + nq * 64 + nt * 16 + ocol;
#pragma unroll
    for (int v = 0; v < 8; ++v)
      op[(size_t)v * 256] = acc[nt][v];
  }
}

// ---------------------------------------------------------------------------
extern "C" void kernel_launch(void* const* d_in, const int* in_sizes, int n_in,
                              void* d_out, int out_size, void* d_ws, size_t ws_size,
                              hipStream_t stream) {
  const float* a       = (const float*)d_in[0];  // [8192,8192]
  const float* x       = (const float*)d_in[1];  // [8192,256,8]
  const float* w_bases = (const float*)d_in[2];  // [4,256,256]
  const float* w_rel   = (const float*)d_in[3];  // [8,4]
  float* out = (float*)d_out;                    // [8192,256]

  unsigned short* Wc   = (unsigned short*)d_ws;                          // 1 MB
  unsigned int*   Yt32 = (unsigned int*)((char*)d_ws + 2048 * 256 * 2);  // 4 MB

  k_combine_w<<<2048, 256, 0, stream>>>(w_bases, w_rel, Wc);
  k_make_yt <<<2048, 256, 0, stream>>>(x, Wc, Yt32);
  k_main_gemm<<<256, 256, 0, stream>>>(a, Yt32, out);
}
